// BitGenModel_49907519979576
// MI455X (gfx1250) — 16-bit and 8-bit matrix operand order repaired and checked by reading, compiled, not run on hardware
//
#include <hip/hip_runtime.h>
#include <hip/hip_bf16.h>
#include <stdint.h>

#define DM    512
#define TLEN  512
#define BSZ   8
#define HN    8
#define HDD   64
#define FFD   2048
#define VSZ   32000
#define NLAY  2
#define MR    (BSZ*TLEN)

#define A_STRIDE   80
#define B_STRIDE   144
#define KT_STRIDE  272
#define KT_HSTR    136
#define V_STRIDE   144
#define V_HSTR     72
#define P_STRIDE   80
#define P_HSTR     40

typedef int      v8i  __attribute__((ext_vector_type(8)));
typedef float    v8f  __attribute__((ext_vector_type(8)));
typedef _Float16 v16h __attribute__((ext_vector_type(16)));

__device__ __forceinline__ v8i wmma_i8(v8i a, v8i b, v8i c) {
  return __builtin_amdgcn_wmma_i32_16x16x64_iu8(true, a, true, b, c, false, false);
}
__device__ __forceinline__ v8f wmma_f16acc(v16h a, v16h b, v8f c) {
  return __builtin_amdgcn_wmma_f32_16x16x32_f16(false, a, false, b, (short)0, c, false, false);
}

__device__ __forceinline__ v16h ldsA16(const char* base, int rowBytes, int row,
                                       int colByte, int hf) {
  const char* p = base + (size_t)row * rowBytes + colByte + hf * 16;
  union { int2 q[4]; v16h vv; } u;
  u.q[0] = *(const int2*)(p + 0);
  u.q[1] = *(const int2*)(p + 8);
  u.q[2] = *(const int2*)(p + 32);
  u.q[3] = *(const int2*)(p + 40);
  return u.vv;
}

__device__ __forceinline__ float redmax16(float v) {
  v = fmaxf(v, __shfl_xor(v, 1, 16));
  v = fmaxf(v, __shfl_xor(v, 2, 16));
  v = fmaxf(v, __shfl_xor(v, 4, 16));
  v = fmaxf(v, __shfl_xor(v, 8, 16));
  return v;
}
__device__ __forceinline__ float redsum16(float v) {
  v += __shfl_xor(v, 1, 16);
  v += __shfl_xor(v, 2, 16);
  v += __shfl_xor(v, 4, 16);
  v += __shfl_xor(v, 8, 16);
  return v;
}

__global__ void abssum_kernel(const float* __restrict__ x, size_t n, float* out) {
  __shared__ float red[256];
  float s = 0.f;
  for (size_t i = (size_t)blockIdx.x * 256 + threadIdx.x; i < n; i += (size_t)gridDim.x * 256)
    s += fabsf(x[i]);
  red[threadIdx.x] = s; __syncthreads();
  for (int st = 128; st > 0; st >>= 1) {
    if ((int)threadIdx.x < st) red[threadIdx.x] += red[threadIdx.x + st];
    __syncthreads();
  }
  if (threadIdx.x == 0) atomicAdd(out, red[0]);
}

__global__ void absmax_kernel(const float* __restrict__ x, size_t n, unsigned* out) {
  __shared__ float red[256];
  float m = 0.f;
  for (size_t i = (size_t)blockIdx.x * 256 + threadIdx.x; i < n; i += (size_t)gridDim.x * 256)
    m = fmaxf(m, fabsf(x[i]));
  red[threadIdx.x] = m; __syncthreads();
  for (int st = 128; st > 0; st >>= 1) {
    if ((int)threadIdx.x < st) red[threadIdx.x] = fmaxf(red[threadIdx.x], red[threadIdx.x + st]);
    __syncthreads();
  }
  if (threadIdx.x == 0) atomicMax(out, __float_as_uint(red[0]));
}

__global__ void quantw_kernel(const float* __restrict__ W, int N, int K,
                              const float* __restrict__ wsum, float invCnt,
                              int8_t* __restrict__ qWt) {
  size_t i = (size_t)blockIdx.x * 256 + threadIdx.x;
  size_t n = (size_t)N * K;
  if (i >= n) return;
  float wscale = fmaxf(wsum[0] * invCnt, 1e-5f);
  float thr = 0.5f * wscale;
  int nn = (int)(i / K), kk = (int)(i % K);
  float w = W[i];
  int8_t q = (fabsf(w) > thr) ? (w > 0.f ? (int8_t)1 : (int8_t)-1) : (int8_t)0;
  qWt[(size_t)kk * N + nn] = q;
}

__global__ void quanta_kernel(const float* __restrict__ x, size_t n,
                              const unsigned* __restrict__ amax, int8_t* __restrict__ q) {
  size_t i = (size_t)blockIdx.x * 256 + threadIdx.x;
  if (i >= n) return;
  float is = fmaxf(__uint_as_float(amax[0]), 1e-8f) * (1.0f / 127.0f);
  float v = rintf(x[i] / is);
  v = fminf(fmaxf(v, -128.f), 127.f);
  q[i] = (int8_t)v;
}

__global__ __launch_bounds__(256) void layernorm_kernel(
    const float* __restrict__ x, const float* __restrict__ g,
    const float* __restrict__ b, float* __restrict__ y) {
  __shared__ float rs[256], rs2[256];
  int row = blockIdx.x, t = threadIdx.x;
  const float* xr = x + (size_t)row * DM;
  float a0 = xr[t], a1 = xr[t + 256];
  rs[t] = a0 + a1; rs2[t] = a0 * a0 + a1 * a1;
  __syncthreads();
  for (int st = 128; st > 0; st >>= 1) {
    if (t < st) { rs[t] += rs[t + st]; rs2[t] += rs2[t + st]; }
    __syncthreads();
  }
  float mean = rs[0] * (1.0f / DM);
  float var  = rs2[0] * (1.0f / DM) - mean * mean;
  float r = rsqrtf(var + 1e-5f);
  float* yr = y + (size_t)row * DM;
  yr[t]       = (a0 - mean) * r * g[t]       + b[t];
  yr[t + 256] = (a1 - mean) * r * g[t + 256] + b[t + 256];
}

__global__ void ctxmean_kernel(const float* __restrict__ enc, float* __restrict__ ctx) {
  int i = blockIdx.x * 256 + threadIdx.x;
  int b = i >> 9, d = i & 511;
  const float* p = enc + ((size_t)b * 256) * DM + d;
  float s = 0.f;
  for (int sIdx = 0; sIdx < 256; ++sIdx) s += p[(size_t)sIdx * DM];
  ctx[i] = s * (1.0f / 256.0f);
}

__global__ void encproj_kernel(const int8_t* __restrict__ qctx, const int8_t* __restrict__ qWt,
                               const float* __restrict__ bias, const float* __restrict__ wsum,
                               float invCnt, const unsigned* __restrict__ amax,
                               float* __restrict__ e) {
  int i = blockIdx.x * 256 + threadIdx.x;
  int b = i >> 9, n = i & 511;
  const int8_t* xq = qctx + b * DM;
  int acc = 0;
  for (int k = 0; k < DM; ++k) acc += (int)xq[k] * (int)qWt[(size_t)k * DM + n];
  float wscale = fmaxf(wsum[0] * invCnt, 1e-5f);
  float is = fmaxf(__uint_as_float(amax[0]), 1e-8f) * (1.0f / 127.0f);
  e[i] = (float)acc * wscale * is + bias[n];
}

template<bool RESID, bool ENC, bool GELU>
__global__ __launch_bounds__(256) void bitgemm_kernel(
    const int8_t* __restrict__ qA, const int8_t* __restrict__ qBt,
    const float* __restrict__ bias, const float* __restrict__ wsum, float invCnt,
    const unsigned* __restrict__ amax,
    const float* resid, const float* enc, float* out,
    int M, int N, int K) {
  (void)M;
  __shared__ __align__(16) int8_t smA[2][128 * A_STRIDE];
  __shared__ __align__(16) int8_t smB[2][64 * B_STRIDE];
  int tid = threadIdx.x, lane = tid & 31, wid = tid >> 5;
  int wm = wid >> 2, wn = wid & 3;
  int hf = lane >> 4, l15 = lane & 15;
  int m0 = blockIdx.y * 128, n0 = blockIdx.x * 128;

  v8i zi = {0, 0, 0, 0, 0, 0, 0, 0};
  v8i acc[4][2];
#pragma unroll
  for (int mt = 0; mt < 4; ++mt)
#pragma unroll
    for (int nt = 0; nt < 2; ++nt) acc[mt][nt] = zi;

  int aIdx = tid * 16;
  int ar0 = aIdx >> 6, ac0 = aIdx & 63;
  int br0 = aIdx >> 7, bc0 = aIdx & 127;

  int4 ra0, ra1, rb0, rb1;
  ra0 = *(const int4*)(qA + (size_t)(m0 + ar0) * K + ac0);
  ra1 = *(const int4*)(qA + (size_t)(m0 + ar0 + 64) * K + ac0);
  rb0 = *(const int4*)(qBt + (size_t)br0 * N + n0 + bc0);
  rb1 = *(const int4*)(qBt + (size_t)(br0 + 32) * N + n0 + bc0);
  *(int4*)(&smA[0][ar0 * A_STRIDE + ac0]) = ra0;
  *(int4*)(&smA[0][(ar0 + 64) * A_STRIDE + ac0]) = ra1;
  *(int4*)(&smB[0][br0 * B_STRIDE + bc0]) = rb0;
  *(int4*)(&smB[0][(br0 + 32) * B_STRIDE + bc0]) = rb1;
  __syncthreads();

  int buf = 0;
  for (int k0 = 0; k0 < K; k0 += 64) {
    bool more = (k0 + 64 < K);
    if (more) {
      ra0 = *(const int4*)(qA + (size_t)(m0 + ar0) * K + k0 + 64 + ac0);
      ra1 = *(const int4*)(qA + (size_t)(m0 + ar0 + 64) * K + k0 + 64 + ac0);
      rb0 = *(const int4*)(qBt + (size_t)(k0 + 64 + br0) * N + n0 + bc0);
      rb1 = *(const int4*)(qBt + (size_t)(k0 + 64 + br0 + 32) * N + n0 + bc0);
    }
    if (k0 + 128 < K) {
      __builtin_prefetch(qA + (size_t)(m0 + ar0) * K + k0 + 128 + ac0, 0, 3);
      __builtin_prefetch(qBt + (size_t)(k0 + 128 + br0) * N + n0 + bc0, 0, 3);
    }
    v8i af[4];
#pragma unroll
    for (int mt = 0; mt < 4; ++mt) {
      const int8_t* p = &smA[buf][(wm * 64 + mt * 16 + l15) * A_STRIDE + hf * 8];
      int2 x0 = *(const int2*)(p + 0);
      int2 x1 = *(const int2*)(p + 16);
      int2 x2 = *(const int2*)(p + 32);
      int2 x3 = *(const int2*)(p + 48);
      v8i a = {x0.x, x0.y, x1.x, x1.y, x2.x, x2.y, x3.x, x3.y};
      af[mt] = a;
    }
    v8i bf[2];
#pragma unroll
    for (int nt = 0; nt < 2; ++nt) {
      const int8_t* p = &smB[buf][(hf * 8) * B_STRIDE + wn * 32 + nt * 16 + l15];
      union { int8_t e[32]; v8i vv; } ub;
#pragma unroll
      for (int s2 = 0; s2 < 32; ++s2) ub.e[s2] = p[((s2 & 7) + ((s2 >> 3) << 4)) * B_STRIDE];
      bf[nt] = ub.vv;
    }
#pragma unroll
    for (int mt = 0; mt < 4; ++mt)
#pragma unroll
      for (int nt = 0; nt < 2; ++nt)
        acc[mt][nt] = wmma_i8(af[mt], bf[nt], acc[mt][nt]);
    __syncthreads();
    if (more) {
      *(int4*)(&smA[buf ^ 1][ar0 * A_STRIDE + ac0]) = ra0;
      *(int4*)(&smA[buf ^ 1][(ar0 + 64) * A_STRIDE + ac0]) = ra1;
      *(int4*)(&smB[buf ^ 1][br0 * B_STRIDE + bc0]) = rb0;
      *(int4*)(&smB[buf ^ 1][(br0 + 32) * B_STRIDE + bc0]) = rb1;
      __syncthreads();
      buf ^= 1;
    }
  }

  float wscale = fmaxf(wsum[0] * invCnt, 1e-5f);
  float iscale = fmaxf(__uint_as_float(amax[0]), 1e-8f) * (1.0f / 127.0f);
  float sc = wscale * iscale;
#pragma unroll
  for (int mt = 0; mt < 4; ++mt)
#pragma unroll
    for (int nt = 0; nt < 2; ++nt)
#pragma unroll
      for (int vv = 0; vv < 8; ++vv) {
        int mrow = m0 + wm * 64 + mt * 16 + vv + hf * 8;
        int ncol = n0 + wn * 32 + nt * 16 + l15;
        float val = (float)acc[mt][nt][vv] * sc + bias[ncol];
        if (GELU) val = 0.5f * val * (1.0f + erff(val * 0.70710678118f));
        if (RESID) val += resid[(size_t)mrow * N + ncol];
        if (ENC) val += enc[((mrow >> 9) << 9) + ncol];
        out[(size_t)mrow * N + ncol] = val;
      }
}

__global__ __launch_bounds__(256) void attention_kernel(
    const float* __restrict__ Qg, const float* __restrict__ Kg,
    const float* __restrict__ Vg, float* __restrict__ Og) {
  __shared__ _Float16 smKt[64 * KT_HSTR];
  __shared__ _Float16 smV[128 * V_HSTR];
  __shared__ _Float16 smP[8 * 16 * P_HSTR];
  int tid = threadIdx.x, lane = tid & 31, wid = tid >> 5;
  int hf = lane >> 4, l15 = lane & 15;
  int b = blockIdx.x >> 3, hh = blockIdx.x & 7;
  int q0 = blockIdx.y * 128;
  size_t baseBH = ((size_t)b * TLEN) * DM + (size_t)hh * HDD;
  int rowg = q0 + wid * 16;

  v16h qa[2];
  {
    const float* qp = Qg + baseBH + (size_t)(rowg + l15) * DM;
#pragma unroll
    for (int c = 0; c < 2; ++c) {
      union { _Float16 e[16]; v16h vv; } u;
#pragma unroll
      for (int vv = 0; vv < 8; ++vv) {
        int ei = c * 32 + (vv & 3) * 2 + hf * 8 + ((vv >> 2) << 4);
        float2 f = *(const float2*)(qp + ei);
        u.e[2 * vv]     = (_Float16)(f.x * 0.125f);
        u.e[2 * vv + 1] = (_Float16)(f.y * 0.125f);
      }
      qa[c] = u.vv;
    }
  }

  float m8[8], l8[8], al8[8];
  v8f zf = {0.f, 0.f, 0.f, 0.f, 0.f, 0.f, 0.f, 0.f};
  v8f o[4];
#pragma unroll
  for (int vv = 0; vv < 8; ++vv) { m8[vv] = -1e30f; l8[vv] = 0.f; }
#pragma unroll
  for (int nt = 0; nt < 4; ++nt) o[nt] = zf;

  for (int kc = 0; kc <= q0 + 127; kc += 128) {
    __syncthreads();
    for (int i = tid; i < 2048; i += 256) {
      int idx = i * 4;
      int r = idx >> 6, hd = idx & 63;
      float4 kf = *(const float4*)(Kg + baseBH + (size_t)(kc + r) * DM + hd);
      smKt[(hd + 0) * KT_HSTR + r] = (_Float16)kf.x;
      smKt[(hd + 1) * KT_HSTR + r] = (_Float16)kf.y;
      smKt[(hd + 2) * KT_HSTR + r] = (_Float16)kf.z;
      smKt[(hd + 3) * KT_HSTR + r] = (_Float16)kf.w;
      float4 vf = *(const float4*)(Vg + baseBH + (size_t)(kc + r) * DM + hd);
      smV[r * V_HSTR + hd + 0] = (_Float16)vf.x;
      smV[r * V_HSTR + hd + 1] = (_Float16)vf.y;
      smV[r * V_HSTR + hd + 2] = (_Float16)vf.z;
      smV[r * V_HSTR + hd + 3] = (_Float16)vf.w;
    }
    __syncthreads();
    int kendw = rowg + 16 - kc;
    if (kendw > 128) kendw = 128;
    for (int kb = 0; kb < kendw; kb += 32) {
      v8f s[2];
#pragma unroll
      for (int jt = 0; jt < 2; ++jt) {
        v8f a = zf;
#pragma unroll
        for (int c = 0; c < 2; ++c) {
          const _Float16* p = smKt + (size_t)(c * 32 + hf * 8) * KT_HSTR + kb + jt * 16 + l15;
          union { _Float16 e[16]; v16h vv; } ub;
#pragma unroll
          for (int s2 = 0; s2 < 16; ++s2) ub.e[s2] = p[(size_t)((s2 & 7) + ((s2 >> 3) << 4)) * KT_HSTR];
          a = wmma_f16acc(qa[c], ub.vv, a);
        }
        s[jt] = a;
      }
      _Float16* pw = smP + wid * 16 * P_HSTR;
#pragma unroll
      for (int vv = 0; vv < 8; ++vv) {
        int row = rowg + vv + hf * 8;
        int c0 = kc + kb + l15, c1 = c0 + 16;
        float s0 = (c0 > row) ? -1e30f : s[0][vv];
        float s1 = (c1 > row) ? -1e30f : s[1][vv];
        float mx = redmax16(fmaxf(s0, s1));
        float mn = fmaxf(m8[vv], mx);
        float e0 = __expf(s0 - mn);
        float e1 = __expf(s1 - mn);
        float rs = redsum16(e0 + e1);
        al8[vv] = __expf(m8[vv] - mn);
        l8[vv] = l8[vv] * al8[vv] + rs;
        m8[vv] = mn;
        pw[(vv + hf * 8) * P_HSTR + l15]      = (_Float16)e0;
        pw[(vv + hf * 8) * P_HSTR + 16 + l15] = (_Float16)e1;
      }
#pragma unroll
      for (int nt = 0; nt < 4; ++nt)
#pragma unroll
        for (int vv = 0; vv < 8; ++vv) o[nt][vv] *= al8[vv];
      v16h pa = ldsA16((const char*)pw, P_STRIDE, l15, 0, hf);
#pragma unroll
      for (int nt = 0; nt < 4; ++nt) {
        const _Float16* p = smV + (size_t)(kb + hf * 8) * V_HSTR + nt * 16 + l15;
        union { _Float16 e[16]; v16h vv; } ub;
#pragma unroll
        for (int s2 = 0; s2 < 16; ++s2) ub.e[s2] = p[(size_t)((s2 & 7) + ((s2 >> 3) << 4)) * V_HSTR];
        o[nt] = wmma_f16acc(pa, ub.vv, o[nt]);
      }
    }
  }
#pragma unroll
  for (int vv = 0; vv < 8; ++vv) {
    float inv = 1.0f / l8[vv];
    size_t rowoff = baseBH + (size_t)(rowg + vv + hf * 8) * DM;
#pragma unroll
    for (int nt = 0; nt < 4; ++nt)
      Og[rowoff + nt * 16 + l15] = o[nt][vv] * inv;
  }
}

extern "C" void kernel_launch(void* const* d_in, const int* in_sizes, int n_in,
                              void* d_out, int out_size, void* d_ws, size_t ws_size,
                              hipStream_t stream) {
  (void)in_sizes; (void)n_in; (void)out_size; (void)ws_size;
  const float* x    = (const float*)d_in[0];
  const float* encO = (const float*)d_in[1];
  const float* Wq   = (const float*)d_in[2];  const float* bq   = (const float*)d_in[3];
  const float* Wk   = (const float*)d_in[4];  const float* bk   = (const float*)d_in[5];
  const float* Wv   = (const float*)d_in[6];  const float* bv   = (const float*)d_in[7];
  const float* Wo   = (const float*)d_in[8];  const float* bo   = (const float*)d_in[9];
  const float* Wenc = (const float*)d_in[10]; const float* benc = (const float*)d_in[11];
  const float* W1   = (const float*)d_in[12]; const float* b1   = (const float*)d_in[13];
  const float* W2   = (const float*)d_in[14]; const float* b2   = (const float*)d_in[15];
  const float* ln1g = (const float*)d_in[16]; const float* ln1b = (const float*)d_in[17];
  const float* ln3g = (const float*)d_in[20]; const float* ln3b = (const float*)d_in[21];
  const float* lnfg = (const float*)d_in[22]; const float* lnfb = (const float*)d_in[23];
  const float* Wout = (const float*)d_in[24]; const float* bout = (const float*)d_in[25];
  float* logits = (float*)d_out;

  char* W = (char*)d_ws;
  size_t off = 0;
  auto carve = [&](size_t bytes) -> char* {
    char* p = W + off; off = (off + bytes + 255) & ~(size_t)255; return p;
  };
  float*    sums  = (float*)carve(256);
  unsigned* amaxs = (unsigned*)((char*)sums + 128);
  int8_t *qWq[NLAY], *qWk[NLAY], *qWv[NLAY], *qWo[NLAY], *qWe[NLAY], *qW1[NLAY], *qW2[NLAY];
  for (int l = 0; l < NLAY; ++l) {
    qWq[l] = (int8_t*)carve((size_t)DM * DM);
    qWk[l] = (int8_t*)carve((size_t)DM * DM);
    qWv[l] = (int8_t*)carve((size_t)DM * DM);
    qWo[l] = (int8_t*)carve((size_t)DM * DM);
    qWe[l] = (int8_t*)carve((size_t)DM * DM);
    qW1[l] = (int8_t*)carve((size_t)DM * FFD);
    qW2[l] = (int8_t*)carve((size_t)FFD * DM);
  }
  int8_t* qWoutT = (int8_t*)carve((size_t)DM * VSZ);
  float*  ctx    = (float*)carve((size_t)BSZ * DM * 4);
  int8_t* qctx   = (int8_t*)carve((size_t)BSZ * DM);
  float*  encb   = (float*)carve((size_t)NLAY * BSZ * DM * 4);
  float*  h      = (float*)carve((size_t)MR * DM * 4);
  float*  y      = (float*)carve((size_t)MR * DM * 4);
  int8_t* qx     = (int8_t*)carve((size_t)MR * FFD);
  float*  Qb     = (float*)carve((size_t)MR * DM * 4 * 4);
  float*  Kb = Qb + (size_t)MR * DM;
  float*  Vb = Kb + (size_t)MR * DM;
  float*  Ob = Vb + (size_t)MR * DM;
  float*  g  = Qb;

  const int SW_Q = 0, SW_K = 2, SW_V = 4, SW_O = 6, SW_E = 8, SW_1 = 10, SW_2 = 12, SW_OUT = 14;
  const int A_CTX = 0, A_YF = 9;
  auto ASLOT = [](int l, int j) { return 1 + l * 4 + j; };

  hipMemsetAsync(sums, 0, 256, stream);

  const float invDD = 1.0f / ((float)DM * DM);
  const float invDF = 1.0f / ((float)DM * FFD);
  const float invDV = 1.0f / ((float)DM * VSZ);

  auto prepW = [&](const float* Wp, int N, int K, int slot, int8_t* qWt, float invCnt) {
    size_t n = (size_t)N * K;
    abssum_kernel<<<256, 256, 0, stream>>>(Wp, n, sums + slot);
    quantw_kernel<<<(unsigned)((n + 255) / 256), 256, 0, stream>>>(Wp, N, K, sums + slot, invCnt, qWt);
  };
  for (int l = 0; l < NLAY; ++l) {
    prepW(Wq + (size_t)l * DM * DM, DM, DM, SW_Q + l, qWq[l], invDD);
    prepW(Wk + (size_t)l * DM * DM, DM, DM, SW_K + l, qWk[l], invDD);
    prepW(Wv + (size_t)l * DM * DM, DM, DM, SW_V + l, qWv[l], invDD);
    prepW(Wo + (size_t)l * DM * DM, DM, DM, SW_O + l, qWo[l], invDD);
    prepW(Wenc + (size_t)l * DM * DM, DM, DM, SW_E + l, qWe[l], invDD);
    prepW(W1 + (size_t)l * FFD * DM, FFD, DM, SW_1 + l, qW1[l], invDF);
    prepW(W2 + (size_t)l * DM * FFD, DM, FFD, SW_2 + l, qW2[l], invDF);
  }
  prepW(Wout, VSZ, DM, SW_OUT, qWoutT, invDV);

  ctxmean_kernel<<<16, 256, 0, stream>>>(encO, ctx);
  absmax_kernel<<<16, 256, 0, stream>>>(ctx, (size_t)BSZ * DM, amaxs + A_CTX);
  quanta_kernel<<<16, 256, 0, stream>>>(ctx, (size_t)BSZ * DM, amaxs + A_CTX, qctx);
  for (int l = 0; l < NLAY; ++l)
    encproj_kernel<<<16, 256, 0, stream>>>(qctx, qWe[l], benc + (size_t)l * DM,
                                           sums + SW_E + l, invDD, amaxs + A_CTX,
                                           encb + (size_t)l * BSZ * DM);

  hipMemcpyAsync(h, x, (size_t)MR * DM * sizeof(float), hipMemcpyDeviceToDevice, stream);

  const size_t nAct = (size_t)MR * DM;
  const size_t nFF  = (size_t)MR * FFD;
  dim3 gD(DM / 128, MR / 128);
  dim3 gF(FFD / 128, MR / 128);
  dim3 gV(VSZ / 128, MR / 128);

  for (int l = 0; l < NLAY; ++l) {
    layernorm_kernel<<<MR, 256, 0, stream>>>(h, ln1g + l * DM, ln1b + l * DM, y);
    absmax_kernel<<<256, 256, 0, stream>>>(y, nAct, amaxs + ASLOT(l, 0));
    quanta_kernel<<<(unsigned)((nAct + 255) / 256), 256, 0, stream>>>(y, nAct, amaxs + ASLOT(l, 0), qx);
    bitgemm_kernel<false, false, false><<<gD, 256, 0, stream>>>(
        qx, qWq[l], bq + l * DM, sums + SW_Q + l, invDD, amaxs + ASLOT(l, 0),
        nullptr, nullptr, Qb, MR, DM, DM);
    bitgemm_kernel<false, false, false><<<gD, 256, 0, stream>>>(
        qx, qWk[l], bk + l * DM, sums + SW_K + l, invDD, amaxs + ASLOT(l, 0),
        nullptr, nullptr, Kb, MR, DM, DM);
    bitgemm_kernel<false, false, false><<<gD, 256, 0, stream>>>(
        qx, qWv[l], bv + l * DM, sums + SW_V + l, invDD, amaxs + ASLOT(l, 0),
        nullptr, nullptr, Vb, MR, DM, DM);
    attention_kernel<<<dim3(BSZ * HN, TLEN / 128), 256, 0, stream>>>(Qb, Kb, Vb, Ob);
    absmax_kernel<<<256, 256, 0, stream>>>(Ob, nAct, amaxs + ASLOT(l, 1));
    quanta_kernel<<<(unsigned)((nAct + 255) / 256), 256, 0, stream>>>(Ob, nAct, amaxs + ASLOT(l, 1), qx);
    bitgemm_kernel<true, true, false><<<gD, 256, 0, stream>>>(
        qx, qWo[l], bo + l * DM, sums + SW_O + l, invDD, amaxs + ASLOT(l, 1),
        h, encb + (size_t)l * BSZ * DM, h, MR, DM, DM);
    layernorm_kernel<<<MR, 256, 0, stream>>>(h, ln3g + l * DM, ln3b + l * DM, y);
    absmax_kernel<<<256, 256, 0, stream>>>(y, nAct, amaxs + ASLOT(l, 2));
    quanta_kernel<<<(unsigned)((nAct + 255) / 256), 256, 0, stream>>>(y, nAct, amaxs + ASLOT(l, 2), qx);
    bitgemm_kernel<false, false, true><<<gF, 256, 0, stream>>>(
        qx, qW1[l], b1 + l * FFD, sums + SW_1 + l, invDF, amaxs + ASLOT(l, 2),
        nullptr, nullptr, g, MR, FFD, DM);
    absmax_kernel<<<256, 256, 0, stream>>>(g, nFF, amaxs + ASLOT(l, 3));
    quanta_kernel<<<(unsigned)((nFF + 255) / 256), 256, 0, stream>>>(g, nFF, amaxs + ASLOT(l, 3), qx);
    bitgemm_kernel<true, false, false><<<gD, 256, 0, stream>>>(
        qx, qW2[l], b2 + l * DM, sums + SW_2 + l, invDF, amaxs + ASLOT(l, 3),
        h, nullptr, h, MR, DM, FFD);
  }

  layernorm_kernel<<<MR, 256, 0, stream>>>(h, lnfg, lnfb, y);
  absmax_kernel<<<256, 256, 0, stream>>>(y, nAct, amaxs + A_YF);
  quanta_kernel<<<(unsigned)((nAct + 255) / 256), 256, 0, stream>>>(y, nAct, amaxs + A_YF, qx);
  bitgemm_kernel<false, false, false><<<gV, 256, 0, stream>>>(
      qx, qWoutT, bout, sums + SW_OUT, invDV, amaxs + A_YF,
      nullptr, nullptr, logits, MR, VSZ, DM);
}
